// AtomTransformer_91027536871991
// MI455X (gfx1250) — matrix operand order repaired and checked by reading, not compiled, not run on hardware
//
#include <hip/hip_runtime.h>
#include <hip/hip_bf16.h>

typedef __attribute__((ext_vector_type(16))) __bf16 v16bf;
typedef __attribute__((ext_vector_type(8)))  float  v8f;
typedef __attribute__((ext_vector_type(4)))  float  f32x4;

#define NTOK 3072
#define CDIM 128

union FragU { v16bf v; f32x4 f4[2]; };

__device__ __forceinline__ v8f zero8() {
  v8f z;
#pragma unroll
  for (int i = 0; i < 8; ++i) z[i] = 0.f;
  return z;
}

__device__ __forceinline__ v8f wmma_bf16(v16bf a, v16bf b, v8f c) {
  // D = A(16x32 bf16) * B(32x16 bf16) + C(16x16 f32)
  return __builtin_amdgcn_wmma_f32_16x16x32_bf16(false, a, false, b, (short)0, c, false, false);
}

// A-fragment: lane row = m0 + lane%16; K = k0 + half*16 + e
__device__ __forceinline__ v16bf ld_afrag(const __bf16* __restrict__ A, int ld, int m0, int k0, int lane) {
  const __bf16* p = A + (size_t)(m0 + (lane & 15)) * ld + k0 + ((lane >> 4) << 4);
  FragU u;
  u.f4[0] = *(const f32x4*)(p);
  u.f4[1] = *(const f32x4*)(p + 8);
  return u.v;
}

// B-fragment from pre-transposed weights Wt[N][K]: lane col = n0 + lane%16; K = k0 + half*16 + e
__device__ __forceinline__ v16bf ld_bfrag(const __bf16* __restrict__ Wt, int ld, int n0, int k0, int lane) {
  const __bf16* p = Wt + (size_t)(n0 + (lane & 15)) * ld + k0 + ((lane >> 4) << 4);
  FragU u;
  u.f4[0] = *(const f32x4*)(p);
  u.f4[1] = *(const f32x4*)(p + 8);
  return u.v;
}

__device__ __forceinline__ float sigf(float x) { return 1.f / (1.f + __expf(-x)); }

__device__ __forceinline__ float wsum(float v) {
#pragma unroll
  for (int o = 16; o > 0; o >>= 1) v += __shfl_xor(v, o, 32);
  return v;
}

// ---------------- generic GEMM: C[m,n] = epi(A[3072xK] @ Wt[NxK]^T) ----------------
template <int K, class Epi>
__global__ __launch_bounds__(256) void gemm_k(const __bf16* __restrict__ A,
                                              const __bf16* __restrict__ Wt,
                                              Epi epi) {
  const int lane = threadIdx.x & 31;
  const int wv = threadIdx.x >> 5;
  const int m0 = blockIdx.x * 128 + wv * 16;
  const int n0 = blockIdx.y * 64;
  v8f acc[4] = {zero8(), zero8(), zero8(), zero8()};
#pragma unroll
  for (int kc = 0; kc < K; kc += 32) {
    v16bf af = ld_afrag(A, K, m0, kc, lane);
#pragma unroll
    for (int nt = 0; nt < 4; ++nt) {
      v16bf bfr = ld_bfrag(Wt, K, n0 + nt * 16, kc, lane);
      acc[nt] = wmma_bf16(af, bfr, acc[nt]);
    }
  }
  const int half = lane >> 4, nl = lane & 15;
#pragma unroll
  for (int nt = 0; nt < 4; ++nt)
#pragma unroll
    for (int r = 0; r < 8; ++r)
      epi(m0 + half * 8 + r, n0 + nt * 16 + nl, acc[nt][r]);
}

// ---------------- epilogues ----------------
struct EpiCond {  // conditioning GEMM: raw + bias, f32 [3072x768]
  float* out; const float* bias;
  __device__ void operator()(int m, int n, float v) const {
    out[(size_t)m * 768 + n] = v + bias[n];
  }
};
struct EpiQKVG {  // cols: [0,128)=q(bf16) [128,256)=k(bf16) [256,384)=v->vT(bf16) [384,512)=g raw f32
  __bf16* q; __bf16* k; __bf16* vT; float* g; const float* bias;
  __device__ void operator()(int m, int n, float v) const {
    v += bias[n];
    if (n < 128)       q[(size_t)m * 128 + n] = (__bf16)v;
    else if (n < 256)  k[(size_t)m * 128 + (n - 128)] = (__bf16)v;
    else if (n < 384)  vT[(size_t)(n - 256) * NTOK + m] = (__bf16)v;
    else               g[(size_t)m * 128 + (n - 384)] = v;
  }
};
struct EpiOut {  // b_out = sigmoid(cond_sg) * (attn @ wo)
  float* bout; const float* cond;
  __device__ void operator()(int m, int n, float v) const {
    bout[(size_t)m * 128 + n] = sigf(cond[(size_t)m * 768 + 256 + n]) * v;
  }
};
struct EpiF32 {  // ff12 raw f32 [3072x512]
  float* out;
  __device__ void operator()(int m, int n, float v) const {
    out[(size_t)m * 512 + n] = v;
  }
};
struct EpiFF3 {  // a_new = b_out + sigmoid(cond_sg2) * (ffh @ ff3)
  float* out; const float* bout; const float* cond;
  __device__ void operator()(int m, int n, float v) const {
    out[(size_t)m * 128 + n] =
        bout[(size_t)m * 128 + n] + sigf(cond[(size_t)m * 768 + 640 + n]) * v;
  }
};

// ---------------- LN(cl) -> bf16 (one wave per row) ----------------
__global__ __launch_bounds__(256) void k_ln_bf16(const float* __restrict__ src,
                                                 __bf16* __restrict__ dst) {
  const int lane = threadIdx.x & 31;
  const int row = blockIdx.x * 8 + (threadIdx.x >> 5);
  const float* r = src + (size_t)row * 128;
  f32x4 v = *(const f32x4*)(r + lane * 4);
  float s = wsum(v.x + v.y + v.z + v.w);
  const float mean = s * (1.f / 128.f);
  f32x4 d = v - mean;
  float ss = wsum(d.x * d.x + d.y * d.y + d.z * d.z + d.w * d.w);
  const float rq = rsqrtf(ss * (1.f / 128.f) + 1e-5f);
#pragma unroll
  for (int j = 0; j < 4; ++j)
    dst[(size_t)row * 128 + lane * 4 + j] = (__bf16)(d[j] * rq);
}

// ---------------- x/y = sigmoid(G)*LN(a) + B (adaLN) ----------------
__global__ __launch_bounds__(256) void k_xy(const float* __restrict__ a,
                                            const float* __restrict__ cond,
                                            __bf16* __restrict__ x,
                                            __bf16* __restrict__ y) {
  const int lane = threadIdx.x & 31;
  const int row = blockIdx.x * 8 + (threadIdx.x >> 5);
  const float* ar = a + (size_t)row * 128;
  f32x4 v = *(const f32x4*)(ar + lane * 4);
  float s = wsum(v.x + v.y + v.z + v.w);
  const float mean = s * (1.f / 128.f);
  f32x4 d = v - mean;
  float ss = wsum(d.x * d.x + d.y * d.y + d.z * d.z + d.w * d.w);
  const float rq = rsqrtf(ss * (1.f / 128.f) + 1e-5f);
  const float* cr = cond + (size_t)row * 768;
#pragma unroll
  for (int j = 0; j < 4; ++j) {
    const int c = lane * 4 + j;
    const float ln = d[j] * rq;
    x[(size_t)row * 128 + c] = (__bf16)(sigf(cr[c])       * ln + cr[128 + c]);
    y[(size_t)row * 128 + c] = (__bf16)(sigf(cr[384 + c]) * ln + cr[512 + c]);
  }
}

// ---------------- swiglu combine: ffh = silu(f1)*f2 ----------------
__global__ void k_swiglu(const float* __restrict__ ff, __bf16* __restrict__ out) {
  const int idx = blockIdx.x * 256 + threadIdx.x;  // < 3072*256
  const int m = idx >> 8, c = idx & 255;
  const float f1 = ff[(size_t)m * 512 + c];
  const float f2 = ff[(size_t)m * 512 + 256 + c];
  out[idx] = (__bf16)(f1 * sigf(f1) * f2);
}

// ---------------- weight transpose + f32->bf16: dst[n*K+k] = src[k*Nn+n] ----------------
__global__ void k_convt(const float* __restrict__ src, __bf16* __restrict__ dst, int K, int Nn) {
  const int idx = blockIdx.x * 256 + threadIdx.x;
  if (idx >= K * Nn) return;
  const int k = idx / Nn, n = idx - k * Nn;
  dst[(size_t)n * K + k] = (__bf16)src[idx];
}

// ---------------- bias assembly ----------------
__global__ void k_bias(const float* __restrict__ agb, const float* __restrict__ sgb,
                       const float* __restrict__ a2gb, const float* __restrict__ sg2b,
                       const float* __restrict__ bq, float* __restrict__ bc,
                       float* __restrict__ bqk) {
  const int t = blockIdx.x * 256 + threadIdx.x;
  if (t < 768) {
    const int j = t >> 7, c = t & 127;
    float v = 0.f;
    if (j == 0) v = agb[c];
    else if (j == 2) v = sgb[c];
    else if (j == 3) v = a2gb[c];
    else if (j == 5) v = sg2b[c];
    bc[t] = v;
  } else if (t < 1280) {
    const int u = t - 768;
    bqk[u] = (u < 128) ? bq[u] : 0.f;
  }
}

// ---------------- local blocked attention: one WG per 32-query block, wave = head ----------------
__global__ __launch_bounds__(128) void k_attn(const __bf16* __restrict__ qb,
                                              const __bf16* __restrict__ kb,
                                              const __bf16* __restrict__ vT,
                                              const float* __restrict__ gbuf,
                                              const float* __restrict__ plm,
                                              const float* __restrict__ zg,
                                              const float* __restrict__ zbeta,
                                              const float* __restrict__ wz,
                                              __bf16* __restrict__ attn_out) {
  // zbS repacked [h][q][nl][nt] so the softmax reads one contiguous b128 per row
  __shared__ __bf16 zbS[4][32][16][8];  // 32 KB
  __shared__ __bf16 pS[4][32][128];     // 32 KB, row-major p for A-frags
  const int c = blockIdx.x;
  const int q0 = c * 32;
  int kw0 = c * 32 - 48;
  if (kw0 < 0) kw0 = 0;
  if (kw0 > NTOK - 128) kw0 = NTOK - 128;
  const float center = (float)(c * 32) + 15.5f;
  const int tid = threadIdx.x;

  // Phase 1: zb[h][q][k] = (LN_cz(plm[q,k,:])*zg + zbeta) @ wz, local window only
  for (int pr = tid; pr < 32 * 128; pr += 128) {
    const int qq = pr >> 7, kk = pr & 127;
    const float* pp = plm + ((size_t)(q0 + qq) * NTOK + (size_t)(kw0 + kk)) * 16;
    if (pr + 128 < 32 * 128)
      __builtin_prefetch(pp + (size_t)NTOK * 16, 0, 0);
    union { f32x4 q4[4]; float s[16]; } u;
#pragma unroll
    for (int j = 0; j < 4; ++j) u.q4[j] = ((const f32x4*)pp)[j];
    float m = 0.f;
#pragma unroll
    for (int j = 0; j < 16; ++j) m += u.s[j];
    m *= (1.f / 16.f);
    float var = 0.f;
#pragma unroll
    for (int j = 0; j < 16; ++j) { const float d = u.s[j] - m; var += d * d; }
    const float rq = rsqrtf(var * (1.f / 16.f) + 1e-5f);
    float hv[4] = {0.f, 0.f, 0.f, 0.f};
#pragma unroll
    for (int j = 0; j < 16; ++j) {
      const float nv = (u.s[j] - m) * rq * zg[j] + zbeta[j];
#pragma unroll
      for (int hh = 0; hh < 4; ++hh) hv[hh] += nv * wz[j * 4 + hh];
    }
    const int wnl = kk & 15, wnt = kk >> 4;
#pragma unroll
    for (int hh = 0; hh < 4; ++hh) zbS[hh][qq][wnl][wnt] = (__bf16)hv[hh];
  }
  __syncthreads();

  const int lane = tid & 31, h = tid >> 5;
  const int half = lane >> 4, nl = lane & 15;
  const float scale = 0.17677669529663689f;  // 1/sqrt(32)

  // block mask depends only on (nt, nl): precompute per-lane
  bool allowed[8];
#pragma unroll
  for (int nt = 0; nt < 8; ++nt)
    allowed[nt] = fabsf((float)(kw0 + nt * 16 + nl) - center) < 64.f;

#pragma unroll 1
  for (int mt = 0; mt < 2; ++mt) {
    // logits = q @ k^T  (A = q[16x32], B = k^T[32x16] per tile; DH=32 = one K-step)
    v16bf aq = ld_afrag(qb + h * 32, CDIM, q0 + mt * 16, 0, lane);
    v8f lg[8];
#pragma unroll
    for (int nt = 0; nt < 8; ++nt) {
      const __bf16* kp = kb + (size_t)(kw0 + nt * 16 + nl) * CDIM + h * 32 + half * 16;
      FragU ub; ub.f4[0] = *(const f32x4*)kp; ub.f4[1] = *(const f32x4*)(kp + 8);
      lg[nt] = wmma_bf16(aq, ub.v, zero8());
    }
    // register-resident softmax: row = half*8+r lives on 16 column-lanes
#pragma unroll
    for (int r = 0; r < 8; ++r) {
      const int mrow = mt * 16 + half * 8 + r;
      union { f32x4 f4; __bf16 b[8]; } zrow;
      zrow.f4 = *(const f32x4*)&zbS[h][mrow][nl][0];  // one ds_load_b128
      float vv[8];
      float mx = -3.0e38f;
#pragma unroll
      for (int nt = 0; nt < 8; ++nt) {
        float val = allowed[nt] ? (lg[nt][r] * scale + (float)zrow.b[nt]) : -3.0e38f;
        vv[nt] = val;
        mx = fmaxf(mx, val);
      }
      mx = fmaxf(mx, __shfl_xor(mx, 1, 32));
      mx = fmaxf(mx, __shfl_xor(mx, 2, 32));
      mx = fmaxf(mx, __shfl_xor(mx, 4, 32));
      mx = fmaxf(mx, __shfl_xor(mx, 8, 32));
      float s = 0.f;
#pragma unroll
      for (int nt = 0; nt < 8; ++nt) { vv[nt] = __expf(vv[nt] - mx); s += vv[nt]; }
      s += __shfl_xor(s, 1, 32);
      s += __shfl_xor(s, 2, 32);
      s += __shfl_xor(s, 4, 32);
      s += __shfl_xor(s, 8, 32);
      const float inv = 1.f / s;
#pragma unroll
      for (int nt = 0; nt < 8; ++nt)
        pS[h][mrow][nt * 16 + nl] = (__bf16)(vv[nt] * inv);
    }
    // o = p @ v : A from LDS pS (row-major), B from vT[d][key] (contiguous keys)
    v8f oacc[2] = {zero8(), zero8()};
#pragma unroll
    for (int kc = 0; kc < 4; ++kc) {
      const __bf16* prow = &pS[h][mt * 16 + nl][kc * 32 + half * 16];
      FragU ua; ua.f4[0] = *(const f32x4*)prow; ua.f4[1] = *(const f32x4*)(prow + 8);
#pragma unroll
      for (int n2 = 0; n2 < 2; ++n2) {
        const __bf16* vp = vT + (size_t)(h * 32 + n2 * 16 + nl) * NTOK + kw0 + kc * 32 + half * 16;
        FragU ub; ub.f4[0] = *(const f32x4*)vp; ub.f4[1] = *(const f32x4*)(vp + 8);
        oacc[n2] = wmma_bf16(ua.v, ub.v, oacc[n2]);
      }
    }
#pragma unroll
    for (int n2 = 0; n2 < 2; ++n2)
#pragma unroll
      for (int r = 0; r < 8; ++r) {
        const int row = q0 + mt * 16 + half * 8 + r;
        const int col = h * 32 + n2 * 16 + nl;
        const float g = gbuf[(size_t)row * CDIM + col];
        attn_out[(size_t)row * CDIM + col] = (__bf16)(oacc[n2][r] * sigf(g));
      }
  }
}

// ---------------- host orchestration ----------------
extern "C" void kernel_launch(void* const* d_in, const int* in_sizes, int n_in,
                              void* d_out, int out_size, void* d_ws, size_t ws_size,
                              hipStream_t stream) {
  (void)in_sizes; (void)n_in; (void)out_size; (void)ws_size;
  const float* ql      = (const float*)d_in[0];
  const float* cl      = (const float*)d_in[1];
  const float* plm     = (const float*)d_in[2];
  const float* ada_g_w = (const float*)d_in[3];
  const float* ada_g_b = (const float*)d_in[4];
  const float* ada_b_w = (const float*)d_in[5];
  const float* wq      = (const float*)d_in[6];
  const float* bq      = (const float*)d_in[7];
  const float* wk      = (const float*)d_in[8];
  const float* wvp     = (const float*)d_in[9];
  const float* z_ln_g  = (const float*)d_in[10];
  const float* z_ln_b  = (const float*)d_in[11];
  const float* wz      = (const float*)d_in[12];
  const float* wg      = (const float*)d_in[13];
  const float* wo      = (const float*)d_in[14];
  const float* sg_w    = (const float*)d_in[15];
  const float* sg_b    = (const float*)d_in[16];
  const float* a2g_w   = (const float*)d_in[17];
  const float* a2g_b   = (const float*)d_in[18];
  const float* a2b_w   = (const float*)d_in[19];
  const float* ff1     = (const float*)d_in[20];
  const float* ff2     = (const float*)d_in[21];
  const float* ff3     = (const float*)d_in[22];
  const float* sg2_w   = (const float*)d_in[23];
  const float* sg2_b   = (const float*)d_in[24];

  char* base = (char*)d_ws;
  size_t off = 0;
  auto alloc = [&](size_t b) { void* p = base + off; off += (b + 255) & ~(size_t)255; return p; };
  __bf16* s_ln   = (__bf16*)alloc((size_t)NTOK * 128 * 2);
  __bf16* WtCond = (__bf16*)alloc((size_t)768 * 128 * 2);
  __bf16* WtQKVG = (__bf16*)alloc((size_t)512 * 128 * 2);
  __bf16* WtO    = (__bf16*)alloc((size_t)128 * 128 * 2);
  __bf16* WtF12  = (__bf16*)alloc((size_t)512 * 128 * 2);
  __bf16* WtF3   = (__bf16*)alloc((size_t)128 * 256 * 2);
  float*  bCond  = (float*)alloc(768 * 4);
  float*  bQKVG  = (float*)alloc(512 * 4);
  float*  condO  = (float*)alloc((size_t)NTOK * 768 * 4);
  __bf16* xb     = (__bf16*)alloc((size_t)NTOK * 128 * 2);
  __bf16* yb     = (__bf16*)alloc((size_t)NTOK * 128 * 2);
  __bf16* qbuf   = (__bf16*)alloc((size_t)NTOK * 128 * 2);
  __bf16* kbuf   = (__bf16*)alloc((size_t)NTOK * 128 * 2);
  __bf16* vTb    = (__bf16*)alloc((size_t)NTOK * 128 * 2);
  float*  gbuf   = (float*)alloc((size_t)NTOK * 128 * 4);
  __bf16* attnb  = (__bf16*)alloc((size_t)NTOK * 128 * 2);
  float*  bout   = (float*)alloc((size_t)NTOK * 128 * 4);
  float*  ff12   = (float*)alloc((size_t)NTOK * 512 * 4);
  __bf16* ffh    = (__bf16*)alloc((size_t)NTOK * 256 * 2);
  float*  abuf0  = (float*)alloc((size_t)NTOK * 128 * 4);
  float*  abuf1  = (float*)alloc((size_t)NTOK * 128 * 4);

  k_ln_bf16<<<NTOK / 8, 256, 0, stream>>>(cl, s_ln);

  auto convt = [&](const float* src, __bf16* dst, int K, int Nn) {
    const int tot = K * Nn;
    k_convt<<<(tot + 255) / 256, 256, 0, stream>>>(src, dst, K, Nn);
  };

  for (int i = 0; i < 3; ++i) {
    const size_t w128 = (size_t)i * 128 * 128;
    const size_t w256 = (size_t)i * 128 * 256;
    convt(ada_g_w + w128, WtCond + 0 * 128 * 128, 128, 128);
    convt(ada_b_w + w128, WtCond + 1 * 128 * 128, 128, 128);
    convt(sg_w    + w128, WtCond + 2 * 128 * 128, 128, 128);
    convt(a2g_w   + w128, WtCond + 3 * 128 * 128, 128, 128);
    convt(a2b_w   + w128, WtCond + 4 * 128 * 128, 128, 128);
    convt(sg2_w   + w128, WtCond + 5 * 128 * 128, 128, 128);
    convt(wq  + w128, WtQKVG + 0 * 128 * 128, 128, 128);
    convt(wk  + w128, WtQKVG + 1 * 128 * 128, 128, 128);
    convt(wvp + w128, WtQKVG + 2 * 128 * 128, 128, 128);
    convt(wg  + w128, WtQKVG + 3 * 128 * 128, 128, 128);
    convt(wo  + w128, WtO, 128, 128);
    convt(ff1 + w256, WtF12, 128, 256);
    convt(ff2 + w256, WtF12 + 256 * 128, 128, 256);
    convt(ff3 + w256, WtF3, 256, 128);
    k_bias<<<5, 256, 0, stream>>>(ada_g_b + i * 128, sg_b + i * 128, a2g_b + i * 128,
                                  sg2_b + i * 128, bq + i * 128, bCond, bQKVG);

    const float* a_in = (i == 0) ? ql : ((i == 1) ? abuf0 : abuf1);
    float* a_out = (i == 2) ? (float*)d_out : ((i == 0) ? abuf0 : abuf1);

    gemm_k<128, EpiCond><<<dim3(NTOK / 128, 12), 256, 0, stream>>>(s_ln, WtCond, EpiCond{condO, bCond});
    k_xy<<<NTOK / 8, 256, 0, stream>>>(a_in, condO, xb, yb);
    gemm_k<128, EpiQKVG><<<dim3(NTOK / 128, 8), 256, 0, stream>>>(xb, WtQKVG,
                                                                  EpiQKVG{qbuf, kbuf, vTb, gbuf, bQKVG});
    k_attn<<<NTOK / 32, 128, 0, stream>>>(qbuf, kbuf, vTb, gbuf, plm,
                                          z_ln_g + i * 16, z_ln_b + i * 16, wz + i * 64, attnb);
    gemm_k<128, EpiOut><<<dim3(NTOK / 128, 2), 256, 0, stream>>>(attnb, WtO, EpiOut{bout, condO});
    gemm_k<128, EpiF32><<<dim3(NTOK / 128, 8), 256, 0, stream>>>(yb, WtF12, EpiF32{ff12});
    k_swiglu<<<(NTOK * 256) / 256, 256, 0, stream>>>(ff12, ffh);
    gemm_k<256, EpiFF3><<<dim3(NTOK / 128, 2), 256, 0, stream>>>(ffh, WtF3, EpiFF3{a_out, bout, condO});
  }
}
